// CapNet_82901458747419
// MI455X (gfx1250) — matrix operand order repaired and checked by reading, not compiled, not run on hardware
//
#include <hip/hip_runtime.h>
#include <hip/hip_bf16.h>
#include <stdint.h>

// ---------------- CDNA5 WMMA / TDM types ----------------
typedef __attribute__((ext_vector_type(16))) _Float16 v16h;
typedef __attribute__((ext_vector_type(8)))  float    v8f;
typedef unsigned int u32x4 __attribute__((ext_vector_type(4)));
typedef int          i32x4 __attribute__((ext_vector_type(4)));
typedef int          i32x8 __attribute__((ext_vector_type(8)));

#if defined(__has_builtin)
# if __has_builtin(__builtin_amdgcn_tensor_load_to_lds)
#  define USE_TDM 1
# endif
#endif
#ifndef USE_TDM
# define USE_TDM 0
#endif

// ---------------- Problem constants ----------------
#define BATCH   128
#define C1      256          // conv1 out channels
#define H_IN    28
#define H1      20           // conv1 out spatial
#define C2      256          // caps conv out channels
#define KTOT    (256*81)     // 20736  implicit-GEMM K  (order: (r*9+s)*256 + c)
#define M2      (BATCH*36)   // 4608   implicit-GEMM M
#define N2      256          // implicit-GEMM N
#define NPRIM   1152
#define NCLS    10
#define CAPD    16
#define EPSV    1e-9f

// ---------------- Workspace layout (bytes) ----------------
#define OFF_H1     ((size_t)0)                               // _Float16 NHWC [B,20,20,256]  26,214,400
#define OFF_W2     ((size_t)26214400)                        // _Float16 [256][20736]        10,616,832
#define OFF_H2     ((size_t)36831232)                        // float    [B,256,36]           4,718,592
#define OFF_UHAT   ((size_t)41549824)                        // float    [B,1152,10,16]      94,371,840
#define OFF_BIJ    ((size_t)135921664)                       // float    [B,1152,10]          5,898,240
#define OFF_CIJ    ((size_t)141819904)                       // float    [B,1152,10]          5,898,240
#define OFF_VJ     ((size_t)147718144)                       // float    [B,10,16]               81,920
#define OFF_MASK   ((size_t)147800064)                       // float    [B,16]                   8,192
#define OFF_FC1    ((size_t)147808256)                       // float    [B,512]                262,144
#define OFF_FC2    ((size_t)148070400)                       // float    [B,1024]               524,288

static __device__ __forceinline__ float relu_f(float v) { return v > 0.f ? v : 0.f; }

// ============ 1) conv1 -> relu -> f16 NHWC [B,20,20,256] ============
__global__ void k_conv1(const float* __restrict__ x, const float* __restrict__ w,
                        const float* __restrict__ b, _Float16* __restrict__ h1) {
    int id = blockIdx.x * blockDim.x + threadIdx.x;
    if (id >= BATCH * C1 * H1 * H1) return;
    const int o    = id & 255;
    const int rest = id >> 8;
    const int pix  = rest % (H1 * H1);
    const int bb   = rest / (H1 * H1);
    const int oy = pix / H1, ox = pix % H1;
    const float* xp = x + (size_t)bb * H_IN * H_IN;
    const float* wp = w + (size_t)o * 81;
    float acc = b[o];
#pragma unroll
    for (int r = 0; r < 9; ++r)
#pragma unroll
        for (int s = 0; s < 9; ++s)
            acc += xp[(oy + r) * H_IN + (ox + s)] * wp[r * 9 + s];
    h1[id] = (_Float16)relu_f(acc);
}

// ============ 2) caps1 weights f32 -> f16, K re-ordered to (r*9+s)*256 + c ============
__global__ void k_cvt_w(const float* __restrict__ w, _Float16* __restrict__ wh) {
    int id = blockIdx.x * blockDim.x + threadIdx.x;
    if (id >= C2 * KTOT) return;
    const int o  = id / KTOT;
    const int k  = id - o * KTOT;
    const int rs = k >> 8;       // 0..80
    const int c  = k & 255;
    wh[id] = (_Float16)w[((size_t)o * 256 + c) * 81 + rs];
}

// ============ 3) caps1 conv as implicit GEMM with WMMA f16 + TDM B-tile ============
// C[M2][N2] = A[M2][KTOT] * B[KTOT][N2]; A = NHWC im2col (contiguous per (r,s)),
// B = re-ordered weights [N][K]. Block tile 128x64, BK=64, 8 waves of 32x32 (4 accs).
// B tile is DMA'd by the Tensor Data Mover straight into LDS (TENSORcnt-tracked).
__global__ __launch_bounds__(256) void k_caps_conv_wmma(
        const _Float16* __restrict__ h1, const _Float16* __restrict__ w2,
        const float* __restrict__ bias, float* __restrict__ h2) {
    constexpr int BM = 128, BN = 64, BK = 64;
    __shared__ _Float16 smem[(BM + BN) * BK];        // sA @ 0, sB @ byte 16384
    _Float16* sA = smem;
    _Float16* sB = smem + BM * BK;
    constexpr unsigned SB_LDS_OFF = BM * BK * sizeof(_Float16);   // 16384

    const int tid  = threadIdx.x;
    const int lane = tid & 31;
    const int wave = tid >> 5;        // 0..7
    const int wm   = wave & 3;        // M rows wm*32 .. +31
    const int wn   = wave >> 2;       // N cols wn*32 .. +31
    const int half = lane >> 4;
    const int l16  = lane & 15;

    const int m0 = blockIdx.y * BM;
    const int n0 = blockIdx.x * BN;

    // A loader: row alm = tid>>1 (0..127), 32 contiguous f16 at offset (tid&1)*32
    const int alm = tid >> 1;
    const int alk = (tid & 1) * 32;
    const int mg  = m0 + alm;
    const int bb  = mg / 36;
    const int pix = mg % 36;
    const int oy2 = (pix / 6) * 2;
    const int ox2 = (pix % 6) * 2;
    const _Float16* hb = h1 + (size_t)bb * (H1 * H1) * 256;

#if USE_TDM
    // D# group1: data_size=2B | tensor_dim0=20736 | tensor_dim1=256 |
    //            tile_dim0=64 | tile_dim1=64 | tensor_dim0_stride=20736
    const i32x8 g1c = { (int)0x00010000u,          // wg_mask=0, data_size=1(2B)
                        (int)0x51000000u,          // tensor_dim0[15:0]=20736 in [31:16]
                        (int)0x01000000u,          // tensor_dim1[15:0]=256 in [31:16]
                        (int)0x00400000u,          // tile_dim0=64 in [31:16]
                        64,                        // tile_dim1=64 (tile_dim2=0)
                        (int)KTOT,                 // tensor_dim0_stride low32
                        0, 0 };                    // stride hi / dim1_stride
    const i32x4 g2z = { 0, 0, 0, 0 };
    const i32x4 g3z = { 0, 0, 0, 0 };
# if defined(__clang_major__) && (__clang_major__ >= 23)
    const i32x8 g5z = { 0, 0, 0, 0, 0, 0, 0, 0 };
# endif
#else
    // fallback manual B loader: row bln = tid>>2 (0..63), 16 f16 at (tid&3)*16
    const int bln = tid >> 2;
    const int blk = (tid & 3) * 16;
    const _Float16* wrow = w2 + (size_t)(n0 + bln) * KTOT;
#endif

    v8f acc[2][2] = {};

    for (int kt = 0; kt < KTOT; kt += BK) {
        // uniform (r,s) for this K-tile: K = (r*9+s)*256 + c, BK divides 256
        const int rs = kt >> 8;
        const int rr = rs / 9;
        const int ss = rs - rr * 9;
        const int cb = kt & 255;

        // ---- A: global -> regs ----
        const uint4* ap = reinterpret_cast<const uint4*>(
            hb + ((oy2 + rr) * H1 + (ox2 + ss)) * 256 + cb + alk);
        const uint4 av0 = ap[0], av1 = ap[1], av2 = ap[2], av3 = ap[3];
        __builtin_prefetch(ap + 4, 0, 1);          // global_prefetch_b8 (next strip)
#if !USE_TDM
        const uint4* bp = reinterpret_cast<const uint4*>(wrow + kt + blk);
        const uint4 bv0 = bp[0], bv1 = bp[1];
#endif

        __syncthreads();   // previous stage done reading LDS

#if USE_TDM
        if (tid < 32) {    // one wave issues the DMA for the whole block
            const unsigned long long ga =
                (unsigned long long)(uintptr_t)(w2 + (size_t)n0 * KTOT + kt);
            const u32x4 g0 = { 1u,                                   // count=1
                               SB_LDS_OFF,                           // lds_addr
                               (unsigned)ga,                         // global_addr lo
                               ((unsigned)(ga >> 32) & 0x01FFFFFFu)  // global_addr hi
                                   | (2u << 30) };                   // type=2 (image)
# if defined(__clang_major__) && (__clang_major__ >= 23)
            __builtin_amdgcn_tensor_load_to_lds(g0, g1c, g2z, g3z, g5z, 0);
# else
            __builtin_amdgcn_tensor_load_to_lds(g0, g1c, g2z, g3z, 0);
# endif
        }
#endif
        {   // A regs -> LDS
            uint4* as = reinterpret_cast<uint4*>(sA + alm * BK + alk);
            as[0] = av0; as[1] = av1; as[2] = av2; as[3] = av3;
        }
#if !USE_TDM
        {
            uint4* bs = reinterpret_cast<uint4*>(sB + bln * BK + blk);
            bs[0] = bv0; bs[1] = bv1;
        }
#else
        if (tid < 32)
            __builtin_amdgcn_s_wait_tensorcnt(0);  // TDM tile landed in LDS
#endif
        __syncthreads();

        // ---- fragments in contiguous k order k = 16*half + s (wave32), 2 K-steps of 32 ----
        const unsigned* A32 = reinterpret_cast<const unsigned*>(sA);
        const unsigned* B32 = reinterpret_cast<const unsigned*>(sB);
#pragma unroll
        for (int ks = 0; ks < BK; ks += 32) {
            union { v16h v; unsigned u[8]; } fa[2], fb[2];
#pragma unroll
            for (int mi = 0; mi < 2; ++mi) {
                const int am = wm * 32 + mi * 16 + l16;
#pragma unroll
                for (int p = 0; p < 8; ++p) {
                    const int ak = p + 8 * half;  // K-pair idx
                    fa[mi].u[p] = A32[am * (BK / 2) + (ks >> 1) + ak];
                }
            }
#pragma unroll
            for (int ni = 0; ni < 2; ++ni) {
                const int bn = wn * 32 + ni * 16 + l16;
#pragma unroll
                for (int p = 0; p < 8; ++p)
                    fb[ni].u[p] = B32[bn * (BK / 2) + (ks >> 1) + 8 * half + p];
            }
#pragma unroll
            for (int mi = 0; mi < 2; ++mi)
#pragma unroll
                for (int ni = 0; ni < 2; ++ni)
                    acc[mi][ni] = __builtin_amdgcn_wmma_f32_16x16x32_f16(
                        false, fa[mi].v, false, fb[ni].v, (short)0,
                        acc[mi][ni], false, false);
        }
    }

    // ---- epilogue: C layout VGPR r -> M = r + 8*half, N = lane&15 ----
#pragma unroll
    for (int mi = 0; mi < 2; ++mi) {
        const int mbase = m0 + wm * 32 + mi * 16 + 8 * half;
#pragma unroll
        for (int r = 0; r < 8; ++r) {
            const int m  = mbase + r;
            const int b_ = m / 36;
            const int pp = m % 36;
#pragma unroll
            for (int ni = 0; ni < 2; ++ni) {
                const int n = n0 + wn * 32 + ni * 16 + l16;
                h2[((size_t)b_ * C2 + n) * 36 + pp] = relu_f(acc[mi][ni][r] + bias[n]);
            }
        }
    }
}

// ============ 4) u_hat: faithful tile+view gather then 16x8 matvec ============
__global__ void k_uhat(const float* __restrict__ h2, const float* __restrict__ W,
                       float* __restrict__ u_hat) {
    int idx = blockIdx.x * blockDim.x + threadIdx.x;           // (b,i,j)
    if (idx >= BATCH * NPRIM * NCLS) return;
    const int j  = idx % NCLS;
    const int bi = idx / NCLS;
    const int i  = bi % NPRIM;
    const int b  = bi / NPRIM;
    float xv[8];
    const int gbase = i * 80 + j * 8;
#pragma unroll
    for (int c = 0; c < 8; ++c) {
        const int g   = gbase + c;
        const int ch  = g / 360;
        const int rem = g - ch * 360;
        const int p36 = rem % 36;
        xv[c] = h2[((size_t)b * C2 + ch) * 36 + p36];
    }
    const float* Wp = W + (size_t)(i * NCLS + j) * 128;        // reinterpreted [16][8]
    float* up = u_hat + (size_t)idx * CAPD;
#pragma unroll
    for (int r = 0; r < CAPD; ++r) {
        float a = 0.f;
#pragma unroll
        for (int c = 0; c < 8; ++c) a += Wp[r * 8 + c] * xv[c];
        up[r] = a;
    }
}

// ============ 5) routing kernels ============
__global__ void k_zero(float* __restrict__ p, int n) {
    int id = blockIdx.x * blockDim.x + threadIdx.x;
    if (id < n) p[id] = 0.f;
}

__global__ void k_softmax(const float* __restrict__ b_ij, float* __restrict__ c_ij) {
    int id = blockIdx.x * blockDim.x + threadIdx.x;            // (b,i)
    if (id >= BATCH * NPRIM) return;
    const float* bp = b_ij + (size_t)id * NCLS;
    float m = -1e30f;
#pragma unroll
    for (int j = 0; j < NCLS; ++j) m = fmaxf(m, bp[j]);
    float e[NCLS], s = 0.f;
#pragma unroll
    for (int j = 0; j < NCLS; ++j) { e[j] = __expf(bp[j] - m); s += e[j]; }
    float inv = 1.f / s;
    float* cp = c_ij + (size_t)id * NCLS;
#pragma unroll
    for (int j = 0; j < NCLS; ++j) cp[j] = e[j] * inv;
}

__global__ __launch_bounds__(128) void k_route_sv(const float* __restrict__ c_ij,
                                                  const float* __restrict__ u_hat,
                                                  const float* __restrict__ caps_bias,
                                                  float* __restrict__ v_j) {
    __shared__ float red[128];
    __shared__ float ssh[16];
    const int b = blockIdx.x / NCLS;
    const int j = blockIdx.x % NCLS;
    const int d = threadIdx.x & 15;
    const int chunk = threadIdx.x >> 4;
    float acc = 0.f;
    for (int ii = 0; ii < 144; ++ii) {
        const int i = chunk * 144 + ii;
        const size_t bij = ((size_t)b * NPRIM + i) * NCLS + j;
        acc += c_ij[bij] * u_hat[bij * CAPD + d];
    }
    red[threadIdx.x] = acc;
    __syncthreads();
    if (threadIdx.x < 16) {
        float s = 0.f;
#pragma unroll
        for (int q = 0; q < 8; ++q) s += red[q * 16 + threadIdx.x];
        s += caps_bias[j * CAPD + threadIdx.x];
        ssh[threadIdx.x] = s;
    }
    __syncthreads();
    if (threadIdx.x < 16) {
        float sq = 0.f;
#pragma unroll
        for (int e = 0; e < 16; ++e) sq += ssh[e] * ssh[e];
        const float sv = ssh[threadIdx.x];
        const float out = (sq / (1.f + sq)) * sv * rsqrtf(sq + EPSV);
        v_j[((size_t)b * NCLS + j) * CAPD + threadIdx.x] = out;
    }
}

__global__ void k_agree(const float* __restrict__ u_hat, const float* __restrict__ v_j,
                        float* __restrict__ b_ij) {
    int idx = blockIdx.x * blockDim.x + threadIdx.x;           // (b,i,j)
    if (idx >= BATCH * NPRIM * NCLS) return;
    const int j = idx % NCLS;
    const int b = idx / (NCLS * NPRIM);
    const float* u = u_hat + (size_t)idx * CAPD;
    const float* v = v_j + ((size_t)b * NCLS + j) * CAPD;
    float a = 0.f;
#pragma unroll
    for (int d = 0; d < CAPD; ++d) a += u[d] * v[d];
    b_ij[idx] += a;
}

// ============ 6) outputs ============
__global__ void k_vlen(const float* __restrict__ v_j, float* __restrict__ out) {
    int id = blockIdx.x * blockDim.x + threadIdx.x;            // (b,j)
    if (id >= BATCH * NCLS) return;
    const float* v = v_j + (size_t)id * CAPD;
    float s = 0.f;
#pragma unroll
    for (int d = 0; d < CAPD; ++d) s += v[d] * v[d];
    out[id] = sqrtf(s + EPSV);
}

__global__ void k_mask(const float* __restrict__ v_j, const float* __restrict__ y,
                       float* __restrict__ masked) {
    int id = blockIdx.x * blockDim.x + threadIdx.x;            // (b,r)
    if (id >= BATCH * 16) return;
    const int r = id % 16, b = id / 16;
    const float* vb = v_j + (size_t)b * 160;
    const float* yb = y + (size_t)b * NCLS;
    float a = 0.f;
#pragma unroll
    for (int c = 0; c < NCLS; ++c) a += vb[r * 10 + c] * yb[c];
    masked[id] = a;
}

__global__ void k_fc1(const float* __restrict__ in, const float* __restrict__ w,
                      const float* __restrict__ b, float* __restrict__ out) {
    int id = blockIdx.x * blockDim.x + threadIdx.x;
    if (id >= BATCH * 512) return;
    const int u = id % 512, bb = id / 512;
    const float* ip = in + (size_t)bb * 16;
    const float* wp = w + (size_t)u * 16;
    float a = b[u];
#pragma unroll
    for (int k = 0; k < 16; ++k) a += ip[k] * wp[k];
    out[id] = a;
}

__global__ void k_fc2(const float* __restrict__ in, const float* __restrict__ w,
                      const float* __restrict__ b, float* __restrict__ out) {
    int id = blockIdx.x * blockDim.x + threadIdx.x;
    if (id >= BATCH * 1024) return;
    const int u = id % 1024, bb = id / 1024;
    const float* ip = in + (size_t)bb * 512;
    const float* wp = w + (size_t)u * 512;
    float a = b[u];
    for (int k = 0; k < 512; ++k) a += ip[k] * wp[k];
    out[id] = a;
}

__global__ void k_fc3_sig(const float* __restrict__ in, const float* __restrict__ w,
                          const float* __restrict__ b, float* __restrict__ out) {
    int id = blockIdx.x * blockDim.x + threadIdx.x;
    if (id >= BATCH * 784) return;
    const int u = id % 784, bb = id / 784;
    const float* ip = in + (size_t)bb * 1024;
    const float* wp = w + (size_t)u * 1024;
    float a = b[u];
    for (int k = 0; k < 1024; ++k) a += ip[k] * wp[k];
    out[(size_t)BATCH * NCLS + (size_t)bb * 784 + u] = 1.f / (1.f + __expf(-a));
}

// ============================ launch ============================
extern "C" void kernel_launch(void* const* d_in, const int* in_sizes, int n_in,
                              void* d_out, int out_size, void* d_ws, size_t ws_size,
                              hipStream_t stream) {
    const float* x       = (const float*)d_in[0];
    const float* y       = (const float*)d_in[1];
    const float* conv1_w = (const float*)d_in[2];
    const float* conv1_b = (const float*)d_in[3];
    const float* caps1_w = (const float*)d_in[4];
    const float* caps1_b = (const float*)d_in[5];
    const float* Wt      = (const float*)d_in[6];
    const float* capsb   = (const float*)d_in[7];
    const float* fc1_w   = (const float*)d_in[8];
    const float* fc1_b   = (const float*)d_in[9];
    const float* fc2_w   = (const float*)d_in[10];
    const float* fc2_b   = (const float*)d_in[11];
    const float* fc3_w   = (const float*)d_in[12];
    const float* fc3_b   = (const float*)d_in[13];
    float* out = (float*)d_out;

    char* ws = (char*)d_ws;
    _Float16* h1   = (_Float16*)(ws + OFF_H1);
    _Float16* w2h  = (_Float16*)(ws + OFF_W2);
    float*    h2   = (float*)(ws + OFF_H2);
    float*    uhat = (float*)(ws + OFF_UHAT);
    float*    bij  = (float*)(ws + OFF_BIJ);
    float*    cij  = (float*)(ws + OFF_CIJ);
    float*    vj   = (float*)(ws + OFF_VJ);
    float*    mskd = (float*)(ws + OFF_MASK);
    float*    f1   = (float*)(ws + OFF_FC1);
    float*    f2   = (float*)(ws + OFF_FC2);

    const int T = 256;

    k_conv1<<<(BATCH * C1 * H1 * H1 + T - 1) / T, T, 0, stream>>>(x, conv1_w, conv1_b, h1);
    k_cvt_w<<<(C2 * KTOT + T - 1) / T, T, 0, stream>>>(caps1_w, w2h);
    k_caps_conv_wmma<<<dim3(N2 / 64, M2 / 128), 256, 0, stream>>>(h1, w2h, caps1_b, h2);
    k_uhat<<<(BATCH * NPRIM * NCLS + T - 1) / T, T, 0, stream>>>(h2, Wt, uhat);
    k_zero<<<(BATCH * NPRIM * NCLS + T - 1) / T, T, 0, stream>>>(bij, BATCH * NPRIM * NCLS);
    for (int r = 0; r < 3; ++r) {
        k_softmax<<<(BATCH * NPRIM + T - 1) / T, T, 0, stream>>>(bij, cij);
        k_route_sv<<<BATCH * NCLS, 128, 0, stream>>>(cij, uhat, capsb, vj);
        if (r < 2)
            k_agree<<<(BATCH * NPRIM * NCLS + T - 1) / T, T, 0, stream>>>(uhat, vj, bij);
    }
    k_vlen<<<(BATCH * NCLS + T - 1) / T, T, 0, stream>>>(vj, out);
    k_mask<<<(BATCH * 16 + T - 1) / T, T, 0, stream>>>(vj, y, mskd);
    k_fc1<<<(BATCH * 512 + T - 1) / T, T, 0, stream>>>(mskd, fc1_w, fc1_b, f1);
    k_fc2<<<(BATCH * 1024 + T - 1) / T, T, 0, stream>>>(f1, fc2_w, fc2_b, f2);
    k_fc3_sig<<<(BATCH * 784 + T - 1) / T, T, 0, stream>>>(f2, fc3_w, fc3_b, out);
}
